// QuadrupletLoss_12378095747855
// MI455X (gfx1250) — matrix operand order repaired and checked by reading, not compiled, not run on hardware
//
#include <hip/hip_runtime.h>
#include <math.h>

// ---------------------------------------------------------------------------
// QuadrupletLoss for MI455X (gfx1250, wave32, WMMA).
// G = X X^T via v_wmma_f32_16x16x32_bf16 with split-bf16 (hi+lo) inputs for
// ~fp32 accuracy: G ~= Ah*Bh + Ah*Bl + Al*Bh.  The 4096x4096 distance matrix
// is never materialized: each wave owns a 32x32 output tile region (2x2
// sub-tiles, A/B fragments reused twice each -> 12 WMMA per 16 b128 loads),
// and fuses the masked row max/min reductions into the tile epilogue.
// Column work is split across blockIdx.y for parallelism; row statistics are
// merged with global u32 atomics on the float bit patterns (valid because
// dist >= 0 makes the IEEE encoding order-isomorphic to unsigned ints).
// ---------------------------------------------------------------------------

typedef __bf16 v16bf __attribute__((ext_vector_type(16)));
typedef float  v8f   __attribute__((ext_vector_type(8)));

#define MARGIN_F      0.3f
#define IDS_PER_SUPER 8
#define POS_INF_BITS  0x7F800000u

union FragBF {
    v16bf v;
    uint4 u[2];   // two 16B halves -> global_load_b128 pairs
};

__device__ __forceinline__ unsigned short f32_to_bf16_rne(float f) {
    unsigned int u = __float_as_uint(f);
    unsigned int lsb = (u >> 16) & 1u;
    u += 0x7fffu + lsb;                 // round to nearest even
    return (unsigned short)(u >> 16);
}
__device__ __forceinline__ float bf16_to_f32(unsigned short h) {
    return __uint_as_float(((unsigned int)h) << 16);
}

// ---------------------------------------------------------------------------
// Kernel 0: seed row statistics (run every launch -> graph-replay safe).
// ap = 0.0f (dist >= 0 so this is a neutral max identity here),
// ani = anc = +inf.  Stored as u32 bit patterns for atomic merging.
// ---------------------------------------------------------------------------
__global__ void init_kernel(unsigned int* __restrict__ ap,
                            unsigned int* __restrict__ ani,
                            unsigned int* __restrict__ anc, int n) {
    const int i = blockIdx.x * 256 + threadIdx.x;
    if (i < n) {
        ap[i]  = 0u;
        ani[i] = POS_INF_BITS;
        anc[i] = POS_INF_BITS;
    }
}

// ---------------------------------------------------------------------------
// Kernel 1: split X (f32) into bf16 hi/lo planes and compute exact row |x|^2.
// One block per row; 256 threads * 4 elements = d (=1024).
// ---------------------------------------------------------------------------
__global__ void prep_kernel(const float* __restrict__ X,
                            unsigned short* __restrict__ Xhi,
                            unsigned short* __restrict__ Xlo,
                            float* __restrict__ sq, int d) {
    __shared__ float red[256];
    const int row = blockIdx.x;
    const int tid = threadIdx.x;

    const float4* xr = (const float4*)(X + (size_t)row * d);
    float4 x = xr[tid];

    float s = x.x * x.x + x.y * x.y + x.z * x.z + x.w * x.w;

    ushort4 hi, lo;
    hi.x = f32_to_bf16_rne(x.x); lo.x = f32_to_bf16_rne(x.x - bf16_to_f32(hi.x));
    hi.y = f32_to_bf16_rne(x.y); lo.y = f32_to_bf16_rne(x.y - bf16_to_f32(hi.y));
    hi.z = f32_to_bf16_rne(x.z); lo.z = f32_to_bf16_rne(x.z - bf16_to_f32(hi.z));
    hi.w = f32_to_bf16_rne(x.w); lo.w = f32_to_bf16_rne(x.w - bf16_to_f32(hi.w));

    *(ushort4*)(Xhi + (size_t)row * d + tid * 4) = hi;
    *(ushort4*)(Xlo + (size_t)row * d + tid * 4) = lo;

    red[tid] = s;
    __syncthreads();
    for (int off = 128; off > 0; off >>= 1) {
        if (tid < off) red[tid] += red[tid + off];
        __syncthreads();
    }
    if (tid == 0) sq[row] = red[0];
}

// ---------------------------------------------------------------------------
// Kernel 2: fused Gram + distance + masked row reductions, 32x32 tile/wave.
// One wave (32 threads, EXEC all-ones as WMMA requires) per (row panel,
// column split).  2x2 sub-tiles of 16x16 reuse each A/B fragment twice.
//
// bf16 A (16x32) lane layout: lane L (lr=L&15, half=L>>4) holds row M=lr,
//   K = {k0+8*half..+7} in v[0..3], K = {k0+16+8*half..+7} in v[4..7].
// bf16 B (32x16) lane layout: lane L holds column N=lr, rows
//   K = {k0+8*half..+7} in v[0..3], K = {k0+16+8*half..+7} in v[4..7]  (B = X^T tile).
// C/D 16x16 f32: acc[r] at lane L = element (row r+8*half, col lr).
// ---------------------------------------------------------------------------
__global__ __launch_bounds__(32)
void gram_dist_kernel(const unsigned short* __restrict__ Xhi,
                      const unsigned short* __restrict__ Xlo,
                      const float* __restrict__ sq,
                      const int* __restrict__ targets,
                      unsigned int* __restrict__ apOut,
                      unsigned int* __restrict__ aniOut,
                      unsigned int* __restrict__ ancOut,
                      int n, int d, int ctPerSplit) {
    const int lane    = threadIdx.x;
    const int half    = lane >> 4;
    const int lr      = lane & 15;
    const int rowBase = blockIdx.x * 32;

    // Per-lane row metadata: rows rowBase + 16*rs + r + 8*half
    float sqi[2][8];
    int   ti[2][8], ci[2][8];
    size_t arow[2];
#pragma unroll
    for (int rs = 0; rs < 2; ++rs) {
        arow[rs] = (size_t)(rowBase + 16 * rs + lr) * d;
#pragma unroll
        for (int r = 0; r < 8; ++r) {
            const int i = rowBase + 16 * rs + r + 8 * half;
            sqi[rs][r] = sq[i];
            ti[rs][r]  = targets[i];
            ci[rs][r]  = (ti[rs][r] - 1) / IDS_PER_SUPER;
        }
    }

    const float INF = __builtin_inff();
    float ap[2][8], ani[2][8], anc[2][8];
#pragma unroll
    for (int rs = 0; rs < 2; ++rs)
#pragma unroll
        for (int r = 0; r < 8; ++r) { ap[rs][r] = 0.0f; ani[rs][r] = INF; anc[rs][r] = INF; }

    const int aOff0 = 8 * half;          // +k0
    const int aOff1 = 16 + 8 * half;     // +k0
    const int bOff0 = 8 * half;          // +k0
    const int bOff1 = 16 + 8 * half;     // +k0

    const int nKT     = d >> 5;
    const int ctStart = blockIdx.y * ctPerSplit;
    const int ctEnd   = ctStart + ctPerSplit;

    for (int ct = ctStart; ct < ctEnd; ++ct) {
        const int colBase = ct * 32;

        float  sqj[2];
        int    tj[2], cj[2];
        size_t brow[2];
#pragma unroll
        for (int cs = 0; cs < 2; ++cs) {
            const int j = colBase + 16 * cs + lr;
            sqj[cs]  = sq[j];
            tj[cs]   = targets[j];
            cj[cs]   = (tj[cs] - 1) / IDS_PER_SUPER;
            brow[cs] = (size_t)j * d;
        }

        // Prefetch next column panel (gfx1250 global_prefetch path).
        if (ct + 1 < ctEnd) {
            const size_t bnext = (size_t)((ct + 1) * 32 + lr) * d;
            __builtin_prefetch(Xhi + bnext, 0, 3);
            __builtin_prefetch(Xlo + bnext, 0, 3);
        }

        v8f acc[2][2];
#pragma unroll
        for (int rs = 0; rs < 2; ++rs)
#pragma unroll
            for (int cs = 0; cs < 2; ++cs) acc[rs][cs] = (v8f){};

#pragma unroll 2
        for (int kt = 0; kt < nKT; ++kt) {
            const int k0 = kt * 32;
            FragBF ah[2], al[2], bh[2], bl[2];
#pragma unroll
            for (int rs = 0; rs < 2; ++rs) {
                ah[rs].u[0] = *(const uint4*)(Xhi + arow[rs] + k0 + aOff0);
                ah[rs].u[1] = *(const uint4*)(Xhi + arow[rs] + k0 + aOff1);
                al[rs].u[0] = *(const uint4*)(Xlo + arow[rs] + k0 + aOff0);
                al[rs].u[1] = *(const uint4*)(Xlo + arow[rs] + k0 + aOff1);
            }
#pragma unroll
            for (int cs = 0; cs < 2; ++cs) {
                bh[cs].u[0] = *(const uint4*)(Xhi + brow[cs] + k0 + bOff0);
                bh[cs].u[1] = *(const uint4*)(Xhi + brow[cs] + k0 + bOff1);
                bl[cs].u[0] = *(const uint4*)(Xlo + brow[cs] + k0 + bOff0);
                bl[cs].u[1] = *(const uint4*)(Xlo + brow[cs] + k0 + bOff1);
            }
            // Split-bf16 product: hi*hi + hi*lo + lo*hi (~fp32 accuracy).
#pragma unroll
            for (int rs = 0; rs < 2; ++rs)
#pragma unroll
                for (int cs = 0; cs < 2; ++cs) {
                    acc[rs][cs] = __builtin_amdgcn_wmma_f32_16x16x32_bf16(
                        false, ah[rs].v, false, bh[cs].v, (short)0, acc[rs][cs], false, false);
                    acc[rs][cs] = __builtin_amdgcn_wmma_f32_16x16x32_bf16(
                        false, ah[rs].v, false, bl[cs].v, (short)0, acc[rs][cs], false, false);
                    acc[rs][cs] = __builtin_amdgcn_wmma_f32_16x16x32_bf16(
                        false, al[rs].v, false, bh[cs].v, (short)0, acc[rs][cs], false, false);
                }
        }

        // Fused epilogue: distance + masked running max/min per row.
#pragma unroll
        for (int rs = 0; rs < 2; ++rs)
#pragma unroll
            for (int cs = 0; cs < 2; ++cs)
#pragma unroll
                for (int r = 0; r < 8; ++r) {
                    const float d2   = sqi[rs][r] + sqj[cs] - 2.0f * acc[rs][cs][r];
                    const float dist = sqrtf(fmaxf(d2, 1e-12f));
                    const bool same_id = (ti[rs][r] == tj[cs]);
                    const bool same_c  = (ci[rs][r] == cj[cs]);
                    if (same_id)      ap[rs][r]  = fmaxf(ap[rs][r],  dist); // positive
                    else if (same_c)  ani[rs][r] = fminf(ani[rs][r], dist); // same class, diff id
                    else              anc[rs][r] = fminf(anc[rs][r], dist); // diff superclass
                }
    }

    // Reduce across the 16 lanes of each half (columns live across lanes).
#pragma unroll
    for (int rs = 0; rs < 2; ++rs)
#pragma unroll
        for (int r = 0; r < 8; ++r) {
#pragma unroll
            for (int off = 8; off > 0; off >>= 1) {
                ap[rs][r]  = fmaxf(ap[rs][r],  __shfl_xor(ap[rs][r],  off, 32));
                ani[rs][r] = fminf(ani[rs][r], __shfl_xor(ani[rs][r], off, 32));
                anc[rs][r] = fminf(anc[rs][r], __shfl_xor(anc[rs][r], off, 32));
            }
        }
    if (lr == 0) {
#pragma unroll
        for (int rs = 0; rs < 2; ++rs)
#pragma unroll
            for (int r = 0; r < 8; ++r) {
                const int i = rowBase + 16 * rs + r + 8 * half;
                // dist >= 0 -> IEEE bits are monotone under unsigned compare.
                atomicMax(&apOut[i],  __float_as_uint(ap[rs][r]));
                atomicMin(&aniOut[i], __float_as_uint(ani[rs][r]));
                atomicMin(&ancOut[i], __float_as_uint(anc[rs][r]));
            }
    }
}

// ---------------------------------------------------------------------------
// Kernel 3: final margin-ranking means -> scalar loss.
// ---------------------------------------------------------------------------
__global__ void loss_kernel(const float* __restrict__ ap,
                            const float* __restrict__ ani,
                            const float* __restrict__ anc,
                            float* __restrict__ out, int n) {
    __shared__ float red[256];
    const int tid = threadIdx.x;
    float s = 0.0f;
    for (int i = tid; i < n; i += 256) {
        const float a = ap[i];
        s += fmaxf(a - ani[i] + MARGIN_F, 0.0f);
        s += fmaxf(a - anc[i] + MARGIN_F, 0.0f);
    }
    red[tid] = s;
    __syncthreads();
    for (int off = 128; off > 0; off >>= 1) {
        if (tid < off) red[tid] += red[tid + off];
        __syncthreads();
    }
    if (tid == 0) out[0] = red[0] / (float)n;   // mean1 + mean2, LOSS_WEIGHT=1
}

// ---------------------------------------------------------------------------
extern "C" void kernel_launch(void* const* d_in, const int* in_sizes, int n_in,
                              void* d_out, int out_size, void* d_ws, size_t ws_size,
                              hipStream_t stream) {
    const float* X       = (const float*)d_in[0];
    const int*   targets = (const int*)d_in[1];

    const int n = in_sizes[1];          // 4096
    const int d = in_sizes[0] / n;      // 1024

    char*  ws  = (char*)d_ws;
    size_t off = 0;
    unsigned short* Xhi = (unsigned short*)(ws + off); off += (size_t)n * d * 2;
    unsigned short* Xlo = (unsigned short*)(ws + off); off += (size_t)n * d * 2;
    float*        sq    = (float*)(ws + off);        off += (size_t)n * 4;
    unsigned int* apB   = (unsigned int*)(ws + off); off += (size_t)n * 4;
    unsigned int* aniB  = (unsigned int*)(ws + off); off += (size_t)n * 4;
    unsigned int* ancB  = (unsigned int*)(ws + off); off += (size_t)n * 4;
    if (ws_size < off) return;  // insufficient scratch

    // Column-split factor: 8 if it divides the 32-wide column tile count.
    const int totalCT = n / 32;
    const int splits  = (totalCT % 8 == 0) ? 8 : 1;
    const int ctPer   = totalCT / splits;

    init_kernel<<<(n + 255) / 256, 256, 0, stream>>>(apB, aniB, ancB, n);
    prep_kernel<<<n, 256, 0, stream>>>(X, Xhi, Xlo, sq, d);
    dim3 grid(n / 32, splits);
    gram_dist_kernel<<<grid, 32, 0, stream>>>(Xhi, Xlo, sq, targets,
                                              apB, aniB, ancB, n, d, ctPer);
    loss_kernel<<<1, 256, 0, stream>>>((const float*)apB, (const float*)aniB,
                                       (const float*)ancB, (float*)d_out, n);
}
